// Qwen3NextGatedDeltaNet_25245817766040
// MI455X (gfx1250) — matrix operand order repaired and checked by reading, not compiled, not run on hardware
//
#include <hip/hip_runtime.h>
#include <hip/hip_bf16.h>
#include <stdint.h>

#define BB 2
#define SS 2048
#define HID_ 2048
#define HK 16
#define HV 32
#define DK 128
#define DV 128
#define KCONV 4
#define KEYD (HK*DK)            // 2048
#define VALD (HV*DV)            // 4096
#define CONVD (2*KEYD+VALD)     // 8192
#define QKVZ_N (2*KEYD+2*VALD)  // 12288
#define BS (BB*SS)              // 4096
#define EPS_ 1e-6f

#if defined(__has_builtin)
#if __has_builtin(__builtin_amdgcn_global_load_async_to_lds_b128)
#define HAS_ASYNC_LDS 1
#else
#define HAS_ASYNC_LDS 0
#endif
#else
#define HAS_ASYNC_LDS 0
#endif

typedef __attribute__((ext_vector_type(16))) __bf16 v16bf;
typedef __attribute__((ext_vector_type(8)))  float  v8f;
typedef int v4i_ __attribute__((vector_size(16)));
typedef __attribute__((address_space(1))) v4i_ v4i_as1;   // global
typedef __attribute__((address_space(3))) v4i_ v4i_as3;   // LDS

union Frag { uint4 u[2]; v16bf v; };

#if HAS_ASYNC_LDS
__device__ __forceinline__ void async_load_b128(const unsigned short* g, unsigned short* l) {
  // global flat address == AS1 address numerically; low 32 bits of an LDS flat
  // address are the LDS byte offset (ISA aperture rule).
  __builtin_amdgcn_global_load_async_to_lds_b128(
      (v4i_as1*)(uintptr_t)g, (v4i_as3*)(unsigned)(uintptr_t)l, 0, 0);
}
#endif

__device__ __forceinline__ void wait_asynccnt4() {
#if HAS_ASYNC_LDS
#if __has_builtin(__builtin_amdgcn_s_wait_asynccnt)
  __builtin_amdgcn_s_wait_asynccnt(4);
#else
  asm volatile("s_wait_asynccnt 4" ::: "memory");
#endif
#endif
}
__device__ __forceinline__ void wait_asynccnt0() {
#if HAS_ASYNC_LDS
#if __has_builtin(__builtin_amdgcn_s_wait_asynccnt)
  __builtin_amdgcn_s_wait_asynccnt(0);
#else
  asm volatile("s_wait_asynccnt 0" ::: "memory");
#endif
#endif
}

__device__ __forceinline__ unsigned short f2bf(float f) {
  unsigned int u = __float_as_uint(f);
  unsigned int r = u + 0x7FFFu + ((u >> 16) & 1u);   // round-to-nearest-even
  return (unsigned short)(r >> 16);
}
__device__ __forceinline__ float sigmoidf_(float x) { return 1.f / (1.f + __expf(-x)); }
__device__ __forceinline__ float siluf_(float x)    { return x * sigmoidf_(x); }
__device__ __forceinline__ float softplusf_(float x){ return (x > 20.f) ? x : log1pf(__expf(x)); }

// ---------------- f32 -> bf16 conversion ----------------
__global__ void k_f32_to_bf16(const float* __restrict__ in, unsigned short* __restrict__ out, int n) {
  int i = (blockIdx.x * blockDim.x + threadIdx.x) * 4;
  if (i + 3 < n) {
    float4 f = *(const float4*)(in + i);
    ushort4 o; o.x = f2bf(f.x); o.y = f2bf(f.y); o.z = f2bf(f.z); o.w = f2bf(f.w);
    *(ushort4*)(out + i) = o;
  } else {
    for (int j = i; j < n; j++) out[j] = f2bf(in[j]);
  }
}

// ---------------- bf16 WMMA GEMM: C[M,N] = A[M,K] * B[N,K]^T ----------------
// M,N multiples of 128; K multiple of 32. 256 threads, 8 waves in 4x2 grid,
// each wave computes 32x64 as 2x4 WMMA 16x16x32 tiles. Double-buffered LDS,
// next tile fetched via global_load_async_to_lds_b128 (ASYNCcnt) when available.
__global__ __launch_bounds__(256) void k_gemm_bf16(
    const unsigned short* __restrict__ A, const unsigned short* __restrict__ Bm,
    float* __restrict__ C, int M, int N, int Kd) {
  (void)M;
  __shared__ __align__(16) unsigned short As[2][128 * 40];
  __shared__ __align__(16) unsigned short Bs[2][128 * 40];
  const int tid = threadIdx.x;
  const int lane = tid & 31;
  const int w = tid >> 5;
  const int wm = w >> 1, wn = w & 1;
  const int blockN = blockIdx.x * 128;
  const int blockM = blockIdx.y * 128;

  v8f acc[2][4];
  #pragma unroll
  for (int i = 0; i < 2; i++)
    #pragma unroll
    for (int j = 0; j < 4; j++) acc[i][j] = (v8f){0.f,0.f,0.f,0.f,0.f,0.f,0.f,0.f};

  const int lr = tid >> 1;         // tile row this thread loads
  const int lc = (tid & 1) * 16;   // 16 bf16 (32B) per thread per tile
  const int m_l = lane & 15, kh = lane >> 4;

  const unsigned short* gA = A  + (size_t)(blockM + lr) * Kd + lc;
  const unsigned short* gB = Bm + (size_t)(blockN + lr) * Kd + lc;
  unsigned short* lA0 = &As[0][lr * 40 + lc];
  unsigned short* lB0 = &Bs[0][lr * 40 + lc];
  const int nk = Kd >> 5;

  auto issue_tile = [&](int it, int buf) {
    const unsigned short* ga = gA + it * 32;
    const unsigned short* gb = gB + it * 32;
    unsigned short* la = lA0 + buf * (128 * 40);
    unsigned short* lb = lB0 + buf * (128 * 40);
#if HAS_ASYNC_LDS
    async_load_b128(ga,     la);
    async_load_b128(ga + 8, la + 8);
    async_load_b128(gb,     lb);
    async_load_b128(gb + 8, lb + 8);
#else
    uint4 a0 = *(const uint4*)ga, a1 = *(const uint4*)(ga + 8);
    uint4 b0 = *(const uint4*)gb, b1 = *(const uint4*)(gb + 8);
    *(uint4*)la = a0; *(uint4*)(la + 8) = a1;
    *(uint4*)lb = b0; *(uint4*)(lb + 8) = b1;
#endif
  };

  issue_tile(0, 0);
  for (int it = 0; it < nk; ++it) {
    const int cur = it & 1;
    if (it + 1 < nk) {
      issue_tile(it + 1, cur ^ 1);
      wait_asynccnt4();            // current tile landed (async loads retire in order)
    } else {
      wait_asynccnt0();
    }
    __syncthreads();               // all waves' share of the tile visible

    Frag af[2], bf[4];
    #pragma unroll
    for (int i = 0; i < 2; i++) {
      int arow = wm * 32 + i * 16 + m_l;
      af[i].u[0] = *(const uint4*)&As[cur][arow * 40 + kh * 8];        // K = kh*8 + 0..7
      af[i].u[1] = *(const uint4*)&As[cur][arow * 40 + 16 + kh * 8];   // K = 16 + kh*8 + 0..7
    }
    #pragma unroll
    for (int j = 0; j < 4; j++) {
      int brow = wn * 64 + j * 16 + m_l;
      const uint4* p = (const uint4*)&Bs[cur][brow * 40 + kh * 8];     // K = kh*8 + 0..7, 16 + kh*8 + 0..7
      bf[j].u[0] = p[0]; bf[j].u[1] = p[2];
    }
    #pragma unroll
    for (int i = 0; i < 2; i++)
      #pragma unroll
      for (int j = 0; j < 4; j++)
        acc[i][j] = __builtin_amdgcn_wmma_f32_16x16x32_bf16(
            false, af[i].v, false, bf[j].v, (short)0, acc[i][j], false, false);

    __syncthreads();               // done reading buf[cur] before it is refilled
  }

  #pragma unroll
  for (int i = 0; i < 2; i++)
    #pragma unroll
    for (int j = 0; j < 4; j++) {
      int col = blockN + wn * 64 + j * 16 + m_l;
      #pragma unroll
      for (int r = 0; r < 8; r++) {
        int row = blockM + wm * 32 + i * 16 + kh * 8 + r;
        C[(size_t)row * N + col] = acc[i][j][r];
      }
    }
}

// ---------------- small ba GEMM: [BS,2048] x [64,2048]^T (fp32) ----------------
__global__ __launch_bounds__(256) void k_ba_gemm(const float* __restrict__ X,
                                                 const float* __restrict__ W,
                                                 float* __restrict__ Y) {
  __shared__ float xr[HID_];
  __shared__ float red[256];
  int row = blockIdx.x;
  for (int j = threadIdx.x; j < HID_; j += 256) xr[j] = X[(size_t)row * HID_ + j];
  __syncthreads();
  int c = threadIdx.x >> 2, part = threadIdx.x & 3;
  const float* wrow = W + (size_t)c * HID_;
  float s = 0.f;
  for (int j = part; j < HID_; j += 4) s = fmaf(xr[j], wrow[j], s);
  red[threadIdx.x] = s;
  __syncthreads();
  if (part == 0)
    Y[(size_t)row * 64 + c] = red[threadIdx.x] + red[threadIdx.x + 1] +
                              red[threadIdx.x + 2] + red[threadIdx.x + 3];
}

// ---------------- gate params g, beta ----------------
__global__ void k_gate(const float* __restrict__ ba, const float* __restrict__ A_log,
                       const float* __restrict__ dt_bias,
                       float* __restrict__ g, float* __restrict__ beta, int n) {
  int idx = blockIdx.x * blockDim.x + threadIdx.x;
  if (idx >= n) return;
  int hv = idx & (HV - 1);
  int bs = idx >> 5;
  int h = hv >> 1, e = hv & 1;
  float av = ba[(size_t)bs * 64 + h * 4 + 2 + e];
  float bv = ba[(size_t)bs * 64 + h * 4 + e];
  g[idx] = -__expf(A_log[hv]) * softplusf_(av + dt_bias[hv]);
  beta[idx] = sigmoidf_(bv);
}

// ---------------- causal depthwise conv (K=4) + SiLU, with qkvz->mixed mapping ----------------
__global__ void k_conv_silu(const float* __restrict__ qkvz, const float* __restrict__ cw,
                            float* __restrict__ qc, float* __restrict__ kc, float* __restrict__ vc) {
  long long idx = (long long)blockIdx.x * blockDim.x + threadIdx.x;  // over BS*CONVD
  int c  = (int)(idx & (CONVD - 1));
  int bt = (int)(idx >> 13);                 // CONVD = 8192 = 2^13
  int t = bt & (SS - 1);
  int b = bt >> 11;                          // SS = 2048 = 2^11
  int src, dstc, dstw; float* dst;
  if (c < KEYD)            { int h = c >> 7, d = c & 127;
                             src = h * 768 + d;                       dst = qc; dstw = KEYD; dstc = c; }
  else if (c < 2 * KEYD)   { int c2 = c - KEYD; int h = c2 >> 7, d = c2 & 127;
                             src = h * 768 + 128 + d;                 dst = kc; dstw = KEYD; dstc = c2; }
  else                     { int c3 = c - 2 * KEYD; int hv = c3 >> 7, dv = c3 & 127;
                             src = (hv >> 1) * 768 + 256 + (hv & 1) * 128 + dv;
                                                                      dst = vc; dstw = VALD; dstc = c3; }
  const float* wp = cw + (size_t)c * KCONV;
  float y = 0.f;
  #pragma unroll
  for (int i = 0; i < KCONV; i++) {
    int tt = t - (KCONV - 1) + i;
    if (tt >= 0) y = fmaf(wp[i], qkvz[(size_t)(b * SS + tt) * QKVZ_N + src], y);
  }
  dst[(size_t)bt * dstw + dstc] = siluf_(y);
}

// ---------------- per-head l2norm for q (with DK^-0.5) and k, in place ----------------
__global__ __launch_bounds__(128) void k_l2norm(float* __restrict__ qc, float* __restrict__ kc) {
  __shared__ float red[128];
  int tid = threadIdx.x;
  size_t base = (size_t)blockIdx.x * DK;     // blocks over BS*HK; (bs,KEYD) contiguous per head
  float q = qc[base + tid];
  red[tid] = q * q; __syncthreads();
  for (int off = 64; off > 0; off >>= 1) { if (tid < off) red[tid] += red[tid + off]; __syncthreads(); }
  float qs = rsqrtf(red[0] + EPS_) * 0.08838834764831845f;  // * DK^-0.5
  __syncthreads();
  float k = kc[base + tid];
  red[tid] = k * k; __syncthreads();
  for (int off = 64; off > 0; off >>= 1) { if (tid < off) red[tid] += red[tid + off]; __syncthreads(); }
  float ks = rsqrtf(red[0] + EPS_);
  qc[base + tid] = q * qs;
  kc[base + tid] = k * ks;
}

// ---------------- gated delta rule scan: one block per (b,hv), thread = one v column ----------------
__global__ __launch_bounds__(128) void k_scan(const float* __restrict__ qn, const float* __restrict__ kn,
                                              const float* __restrict__ vc, const float* __restrict__ g,
                                              const float* __restrict__ beta, float* __restrict__ core) {
  __shared__ __align__(16) float lq[DK];
  __shared__ __align__(16) float lk[DK];
  const int tid = threadIdx.x;
  const int bh = blockIdx.x;       // b*HV + hv
  const int b = bh >> 5;
  const int hv = bh & 31;
  const int h = hv >> 1;
  float s[DK];
  #pragma unroll
  for (int i = 0; i < DK; i++) s[i] = 0.f;
  const float4* lk4 = (const float4*)lk;
  const float4* lq4 = (const float4*)lq;

  for (int t = 0; t < SS; t++) {
    size_t bs = (size_t)(b * SS + t);
    lq[tid] = qn[bs * KEYD + h * DK + tid];
    lk[tid] = kn[bs * KEYD + h * DK + tid];
    float v_t = vc[bs * VALD + hv * DV + tid];
    float g_t = g[bs * HV + hv];
    float b_t = beta[bs * HV + hv];
    __syncthreads();
    float eg = __expf(g_t);
    float kv0 = 0.f, kv1 = 0.f, kv2 = 0.f, kv3 = 0.f;   // 4 chains for ILP
    #pragma unroll
    for (int i = 0; i < DK / 4; i++) {
      float4 kk = lk4[i];
      s[4*i+0] *= eg; s[4*i+1] *= eg; s[4*i+2] *= eg; s[4*i+3] *= eg;
      kv0 = fmaf(kk.x, s[4*i+0], kv0);
      kv1 = fmaf(kk.y, s[4*i+1], kv1);
      kv2 = fmaf(kk.z, s[4*i+2], kv2);
      kv3 = fmaf(kk.w, s[4*i+3], kv3);
    }
    float delta = (v_t - ((kv0 + kv1) + (kv2 + kv3))) * b_t;
    float o0 = 0.f, o1 = 0.f, o2 = 0.f, o3 = 0.f;
    #pragma unroll
    for (int i = 0; i < DK / 4; i++) {
      float4 kk = lk4[i]; float4 qq = lq4[i];
      s[4*i+0] = fmaf(kk.x, delta, s[4*i+0]); o0 = fmaf(qq.x, s[4*i+0], o0);
      s[4*i+1] = fmaf(kk.y, delta, s[4*i+1]); o1 = fmaf(qq.y, s[4*i+1], o1);
      s[4*i+2] = fmaf(kk.z, delta, s[4*i+2]); o2 = fmaf(qq.z, s[4*i+2], o2);
      s[4*i+3] = fmaf(kk.w, delta, s[4*i+3]); o3 = fmaf(qq.w, s[4*i+3], o3);
    }
    core[(bs * HV + hv) * DV + tid] = (o0 + o1) + (o2 + o3);
    __syncthreads();
  }
}

// ---------------- gated RMSNorm + SiLU(z) gate, emit bf16 for the final GEMM ----------------
__global__ __launch_bounds__(128) void k_rmsnorm_gate(const float* __restrict__ core,
                                                      const float* __restrict__ qkvz,
                                                      const float* __restrict__ nw,
                                                      unsigned short* __restrict__ out_bf16) {
  __shared__ float red[128];
  int tid = threadIdx.x;
  int vec = blockIdx.x;            // bs*HV + hv
  int hv = vec & 31;
  int bs = vec >> 5;
  float x = core[(size_t)vec * DV + tid];
  red[tid] = x * x; __syncthreads();
  for (int off = 64; off > 0; off >>= 1) { if (tid < off) red[tid] += red[tid + off]; __syncthreads(); }
  float inv = rsqrtf(red[0] * (1.f / DV) + EPS_);
  float z = qkvz[(size_t)bs * QKVZ_N + (hv >> 1) * 768 + 512 + (hv & 1) * 128 + tid];
  float y = x * inv * nw[tid] * siluf_(z);
  out_bf16[(size_t)bs * VALD + hv * DV + tid] = f2bf(y);
}

extern "C" void kernel_launch(void* const* d_in, const int* in_sizes, int n_in,
                              void* d_out, int out_size, void* d_ws, size_t ws_size,
                              hipStream_t stream) {
  (void)in_sizes; (void)n_in; (void)out_size; (void)ws_size;
  const float* hidden  = (const float*)d_in[0];
  const float* W_qkvz  = (const float*)d_in[1];
  const float* W_ba    = (const float*)d_in[2];
  const float* conv_w  = (const float*)d_in[3];
  const float* dt_bias = (const float*)d_in[4];
  const float* A_log   = (const float*)d_in[5];
  const float* norm_w  = (const float*)d_in[6];
  const float* W_out   = (const float*)d_in[7];
  float* out = (float*)d_out;

  char* w = (char*)d_ws;
  size_t off = 0;
  auto alloc = [&](size_t bytes) -> void* {
    void* p = w + off; off = (off + bytes + 255) & ~(size_t)255; return p;
  };
  unsigned short* hid_bf  = (unsigned short*)alloc((size_t)BS * HID_ * 2);
  unsigned short* wq_bf   = (unsigned short*)alloc((size_t)QKVZ_N * HID_ * 2);
  unsigned short* wo_bf   = (unsigned short*)alloc((size_t)HID_ * VALD * 2);
  float* qkvz             = (float*)alloc((size_t)BS * QKVZ_N * 4);
  float* ba               = (float*)alloc((size_t)BS * 64 * 4);
  float* qc               = (float*)alloc((size_t)BS * KEYD * 4);
  float* kc               = (float*)alloc((size_t)BS * KEYD * 4);
  float* vc               = (float*)alloc((size_t)BS * VALD * 4);
  float* gbuf             = (float*)alloc((size_t)BS * HV * 4);
  float* bbuf             = (float*)alloc((size_t)BS * HV * 4);
  float* core             = (float*)alloc((size_t)BS * VALD * 4);
  unsigned short* core_bf = (unsigned short*)alloc((size_t)BS * VALD * 2);

  // 1) bf16 conversions for the two big GEMMs
  { int n = BS * HID_;     k_f32_to_bf16<<<(n/4 + 255)/256, 256, 0, stream>>>(hidden, hid_bf, n); }
  { int n = QKVZ_N * HID_; k_f32_to_bf16<<<(n/4 + 255)/256, 256, 0, stream>>>(W_qkvz, wq_bf, n); }
  { int n = HID_ * VALD;   k_f32_to_bf16<<<(n/4 + 255)/256, 256, 0, stream>>>(W_out, wo_bf, n); }

  // 2) qkvz = X @ W_qkvz^T   [4096 x 12288]
  k_gemm_bf16<<<dim3(QKVZ_N / 128, BS / 128), 256, 0, stream>>>(hid_bf, wq_bf, qkvz, BS, QKVZ_N, HID_);

  // 3) ba = X @ W_ba^T       [4096 x 64]
  k_ba_gemm<<<BS, 256, 0, stream>>>(hidden, W_ba, ba);

  // 4) gate params g, beta
  k_gate<<<(BS * HV) / 256, 256, 0, stream>>>(ba, A_log, dt_bias, gbuf, bbuf, BS * HV);

  // 5) causal conv + SiLU -> qc, kc, vc
  k_conv_silu<<<(unsigned)(((long long)BS * CONVD) / 256), 256, 0, stream>>>(qkvz, conv_w, qc, kc, vc);

  // 6) l2norm(q)*DK^-0.5, l2norm(k) in place
  k_l2norm<<<BS * HK, 128, 0, stream>>>(qc, kc);

  // 7) gated delta rule recurrence
  k_scan<<<BB * HV, 128, 0, stream>>>(qc, kc, vc, gbuf, bbuf, core);

  // 8) gated RMSNorm (+SiLU(z)) -> bf16
  k_rmsnorm_gate<<<BS * HV, 128, 0, stream>>>(core, qkvz, norm_w, core_bf);

  // 9) out = core @ W_out^T   [4096 x 2048]
  k_gemm_bf16<<<dim3(HID_ / 128, BS / 128), 256, 0, stream>>>(core_bf, wo_bf, out, BS, HID_, VALD);
}
